// Handmade_conv2d_implementation_32280974196851
// MI455X (gfx1250) — matrix operand order repaired and checked by reading, not compiled, not run on hardware
//
#include <hip/hip_runtime.h>

// ---------------------------------------------------------------------------
// Implicit-GEMM 3x3 VALID conv for MI455X (gfx1250, wave32, WMMA).
//   input  [32,128,64,64] f32 (NCHW), weights [256,128,3,3] f32 (OIHW)
//   output [32,256,62,62] f32 (NCHW)
// Strategy: f32 -> f16 layout transforms (NHWC input, [tap][cout][cin] weights),
// then per-wave 32x32 output tile via 2x2 v_wmma_f32_16x16x32_f16 subtiles,
// K = cin (contiguous in memory after transform) looped over 9 taps.
// ---------------------------------------------------------------------------

typedef _Float16 __attribute__((ext_vector_type(8)))  v8h;
typedef _Float16 __attribute__((ext_vector_type(16))) v16h;
typedef float    __attribute__((ext_vector_type(8)))  v8f;

#define CIN   128
#define COUT  256
#define HIN   64
#define WIN   64
#define HOUT  62
#define WOUT  62
#define NBATCH 32
#define PIX_PER_IMG (HOUT * WOUT)          // 3844
#define NPIX (NBATCH * PIX_PER_IMG)        // 123008 = 3844 tiles of 32, exact
#define WT16_ELEMS (9 * COUT * CIN)        // 294912
#define IN16_ELEMS (NBATCH * HIN * WIN * CIN) // 16777216

// ---- NCHW f32 -> NHWC f16 ----------------------------------------------------
// j = ((b*64 + y)*64 + x)*128 + ci  (write fully coalesced; read is a gather,
// one-time 64MB pass, negligible vs. conv)
__global__ __launch_bounds__(256) void cvt_input_nchw_to_nhwc_f16(
    const float* __restrict__ in, _Float16* __restrict__ out) {
  int j  = (int)blockIdx.x * 256 + (int)threadIdx.x;
  int ci = j & (CIN - 1);
  int x  = (j >> 7)  & (WIN - 1);
  int y  = (j >> 13) & (HIN - 1);
  int b  = j >> 19;
  out[j] = (_Float16)in[(((b * CIN + ci) * HIN + y) * WIN) + x];
}

// ---- OIHW f32 -> [tap][cout][cin] f16 ---------------------------------------
__global__ __launch_bounds__(256) void cvt_weights_f16(
    const float* __restrict__ w, _Float16* __restrict__ out) {
  int j   = (int)blockIdx.x * 256 + (int)threadIdx.x;
  int ci  = j & (CIN - 1);
  int co  = (j >> 7) & (COUT - 1);
  int tap = j >> 15;                  // 0..8
  int kh  = tap / 3;
  int kw  = tap - kh * 3;
  out[j] = (_Float16)w[((co * CIN + ci) * 3 + kh) * 3 + kw];
}

// ---- main conv ---------------------------------------------------------------
__global__ __launch_bounds__(256) void conv3x3_implicit_gemm_wmma(
    const _Float16* __restrict__ wt16,   // [9][256][128]
    const _Float16* __restrict__ in16,   // [32][64][64][128]
    float* __restrict__ out)             // [32][256][62][62]
{
  const int lane    = (int)threadIdx.x & 31;
  const int wave    = (int)threadIdx.x >> 5;
  const int wg      = (int)blockIdx.x * 8 + wave;
  const int mtile   = wg & 7;            // 8 cout tiles of 32; waves in a block
  const int ntile   = wg >> 3;           // share the pixel tile -> hot B in L2/L0
  const int cobase  = mtile * 32;
  const int pixbase = ntile * 32;

  const int ln15 = lane & 15;
  const int hi   = lane >> 4;
  // 16-bit operand k order used here (contiguous, not the ISA table order; wave32):
  //  A 16x32: lanes 0-15 hold K 0..15 ; lanes 16-31 hold K 16..31 (matches B)
  //  B 32x16: lane = column n, lanes 0-15 hold K 0..15, lanes 16-31 hold K 16..31
  const int klo  = hi ? 16 : 0;
  const int koff = hi ? 16 : 0;

  // Decode the two pixel sub-tile columns owned by this lane.
  int P0 = pixbase + ln15;
  int P1 = P0 + 16;
  int b0 = P0 / PIX_PER_IMG; int r0 = P0 - b0 * PIX_PER_IMG;
  int y0 = r0 / WOUT;        int x0 = r0 - y0 * WOUT;
  int b1 = P1 / PIX_PER_IMG; int r1 = P1 - b1 * PIX_PER_IMG;
  int y1 = r1 / WOUT;        int x1 = r1 - y1 * WOUT;

  const _Float16* inb0 = in16 + ((size_t)((b0 * HIN + y0) * WIN + x0)) * CIN + koff;
  const _Float16* inb1 = in16 + ((size_t)((b1 * HIN + y1) * WIN + x1)) * CIN + koff;
  const _Float16* wa0  = wt16 + (size_t)(cobase + ln15) * CIN + klo;
  const _Float16* wa1  = wa0 + 16 * CIN;

  v8f acc00 = {}, acc01 = {}, acc10 = {}, acc11 = {};

  for (int tap = 0; tap < 9; ++tap) {
    const int kh = tap / 3, kw = tap - kh * 3;
    const _Float16* pa0 = wa0 + (size_t)tap * (COUT * CIN);
    const _Float16* pa1 = wa1 + (size_t)tap * (COUT * CIN);
    const _Float16* pb0 = inb0 + (kh * WIN + kw) * CIN;
    const _Float16* pb1 = inb1 + (kh * WIN + kw) * CIN;

    // Pull next tap's weight block toward the caches (global_prefetch_b8).
    __builtin_prefetch(pa0 + (COUT * CIN), 0, 3);

#pragma unroll
    for (int kc = 0; kc < CIN; kc += 32) {
      // A fragment: elems 0..15 <- K klo+kc..+15, contiguous
      v8h a0lo = *(const v8h*)(pa0 + kc);
      v8h a0hi = *(const v8h*)(pa0 + kc + 8);
      v8h a1lo = *(const v8h*)(pa1 + kc);
      v8h a1hi = *(const v8h*)(pa1 + kc + 8);
      // B fragment: elems 0..15 <- K koff+kc..+15, one contiguous 32B run
      v8h b0lo = *(const v8h*)(pb0 + kc);
      v8h b0hi = *(const v8h*)(pb0 + kc + 8);
      v8h b1lo = *(const v8h*)(pb1 + kc);
      v8h b1hi = *(const v8h*)(pb1 + kc + 8);

      v16h A0, A1, B0, B1;
#pragma unroll
      for (int i = 0; i < 8; ++i) {
        A0[i] = a0lo[i]; A0[8 + i] = a0hi[i];
        A1[i] = a1lo[i]; A1[8 + i] = a1hi[i];
        B0[i] = b0lo[i]; B0[8 + i] = b0hi[i];
        B1[i] = b1lo[i]; B1[8 + i] = b1hi[i];
      }
      acc00 = __builtin_amdgcn_wmma_f32_16x16x32_f16(false, A0, false, B0, (short)0, acc00, false, false);
      acc01 = __builtin_amdgcn_wmma_f32_16x16x32_f16(false, A0, false, B1, (short)0, acc01, false, false);
      acc10 = __builtin_amdgcn_wmma_f32_16x16x32_f16(false, A1, false, B0, (short)0, acc10, false, false);
      acc11 = __builtin_amdgcn_wmma_f32_16x16x32_f16(false, A1, false, B1, (short)0, acc11, false, false);
    }
  }

  // C/D layout: VGPR v, lanes 0-15 -> row m=v, lanes 16-31 -> row m=v+8; col n=ln15.
  const size_t o0 = (size_t)b0 * COUT * PIX_PER_IMG + (size_t)y0 * WOUT + x0;
  const size_t o1 = (size_t)b1 * COUT * PIX_PER_IMG + (size_t)y1 * WOUT + x1;
#pragma unroll
  for (int v = 0; v < 8; ++v) {
    int m   = v + hi * 8;
    int co0 = cobase + m;
    int co1 = co0 + 16;
    out[o0 + (size_t)co0 * PIX_PER_IMG] = acc00[v];
    out[o1 + (size_t)co0 * PIX_PER_IMG] = acc01[v];
    out[o0 + (size_t)co1 * PIX_PER_IMG] = acc10[v];
    out[o1 + (size_t)co1 * PIX_PER_IMG] = acc11[v];
  }
}

extern "C" void kernel_launch(void* const* d_in, const int* in_sizes, int n_in,
                              void* d_out, int out_size, void* d_ws, size_t ws_size,
                              hipStream_t stream) {
  (void)in_sizes; (void)n_in; (void)out_size; (void)ws_size;
  const float* in = (const float*)d_in[0];
  const float* w  = (const float*)d_in[1];
  float* out      = (float*)d_out;

  _Float16* wt16 = (_Float16*)d_ws;                                  // 589,824 B
  _Float16* in16 = (_Float16*)((char*)d_ws + WT16_ELEMS * sizeof(_Float16)); // +33.55 MB

  cvt_weights_f16<<<WT16_ELEMS / 256, 256, 0, stream>>>(w, wt16);
  cvt_input_nchw_to_nhwc_f16<<<IN16_ELEMS / 256, 256, 0, stream>>>(in, in16);

  // 8 cout-tiles * 3844 pixel-tiles = 30752 waves = 3844 blocks of 8 waves.
  conv3x3_implicit_gemm_wmma<<<(8 * (NPIX / 32)) / 8, 256, 0, stream>>>(wt16, in16, out);
}
